// GATRegression_2903397892408
// MI455X (gfx1250) — matrix operand order repaired and checked by reading, not compiled, not run on hardware
//
#include <hip/hip_runtime.h>

#define N_NODES 50000
#define E_EDGES 800000
#define ETOT (E_EDGES + N_NODES)
#define GRAPHS 64

typedef unsigned short u16;
typedef __attribute__((ext_vector_type(16))) __bf16       bf16x16;
typedef __attribute__((ext_vector_type(8)))  float        f32x8;
typedef __attribute__((ext_vector_type(4)))  unsigned int u32x4;

union FragCast { u32x4 q[2]; bf16x16 v; };

__device__ __forceinline__ u16 f2bf(float f) {
  unsigned int u = __float_as_uint(f);
  u += 0x7fffu + ((u >> 16) & 1u);   // round-to-nearest-even bf16
  return (u16)(u >> 16);
}

__device__ __forceinline__ float leaky(float v) { return v > 0.0f ? v : 0.2f * v; }

__device__ __forceinline__ void edge_endpoints(const int* __restrict__ ei, int e, int& s, int& d) {
  if (e < E_EDGES) { s = ei[e]; d = ei[E_EDGES + e]; }
  else { s = e - E_EDGES; d = s; }   // appended self-loops
}

// float atomic max via signed-max / unsigned-min bit trick
__device__ __forceinline__ void atomicMaxF32(float* addr, float v) {
  if (v >= 0.0f) atomicMax((int*)addr, __float_as_int(v));
  else atomicMin((unsigned int*)addr, (unsigned int)__float_as_int(v));
}

// ---------------------------------------------------------------------------
// bf16 WMMA GEMM, LDS-free: C[M,256] = A[M,K](bf16, row-major) * W[K,256].
// W is pre-transposed to Wt[256][K] so both A and B fragments are two aligned
// global b128 loads per lane, exactly in the ISA 7.12.2 wave32 layouts.
// Block = 128 threads = 4 wave32s; wave w owns the 16x64 strip at col w*64;
// grid.x = M/16 (exact). No divergence -> EXEC all ones for WMMA.
// ---------------------------------------------------------------------------
__global__ __launch_bounds__(128) void gemm_bf16_wmma(
    const u16* __restrict__ A, const u16* __restrict__ Wt,
    float* __restrict__ C, int K) {
  const int t    = threadIdx.x;
  const int m0   = blockIdx.x << 4;
  const int lane = t & 31;
  const int wv   = t >> 5;
  const int hi   = lane >> 4;   // half-wave select
  const int mrow = lane & 15;

  f32x8 acc[4];
#pragma unroll
  for (int i = 0; i < 4; ++i)
#pragma unroll
    for (int r = 0; r < 8; ++r) acc[i][r] = 0.0f;

  // A fragment: halves 0..7 -> K = kk + 8*hi .. +7 ; halves 8..15 -> K = kk+16+8*hi .. +7
  const u16* arow = A + (size_t)(m0 + mrow) * K + (hi << 3);
  // B fragment: halves 0..7 -> K = kk + 8*hi .. +7 ; halves 8..15 -> K = kk+16+8*hi .. +7
  const u16* wcol[4];
#pragma unroll
  for (int nt = 0; nt < 4; ++nt)
    wcol[nt] = Wt + (size_t)((wv << 6) + (nt << 4) + mrow) * K + (hi << 3);

  for (int kk = 0; kk < K; kk += 32) {
    FragCast af;
    af.q[0] = *(const u32x4*)(arow + kk);
    af.q[1] = *(const u32x4*)(arow + kk + 16);
#pragma unroll
    for (int nt = 0; nt < 4; ++nt) {
      FragCast bfr;
      bfr.q[0] = *(const u32x4*)(wcol[nt] + kk);
      bfr.q[1] = *(const u32x4*)(wcol[nt] + kk + 16);
      acc[nt] = __builtin_amdgcn_wmma_f32_16x16x32_bf16(
          false, af.v, false, bfr.v, (short)0, acc[nt], false, false);
    }
  }
  // C layout: VGPR r -> row m0 + r + 8*hi, col = strip + mrow
#pragma unroll
  for (int nt = 0; nt < 4; ++nt) {
    int col = (wv << 6) + (nt << 4) + mrow;
    float* cp = C + (size_t)(m0 + (hi << 3)) * 256 + col;
#pragma unroll
    for (int r = 0; r < 8; ++r) cp[(size_t)r * 256] = acc[nt][r];
  }
}

// ---------------------------------------------------------------------------
// W[K][256] f32 -> Wt[256][K] bf16 (fragment-ready, done once; K*256 <= 65536)
__global__ void w_transpose_bf16_k(const float* __restrict__ W, u16* __restrict__ Wt, int K) {
  int i = blockIdx.x * 256 + threadIdx.x;
  if (i >= K * 256) return;
  int k = i >> 8, n = i & 255;
  Wt[(size_t)n * K + k] = f2bf(W[i]);
}

__global__ void cvt_bf16_k(const float* __restrict__ src, u16* __restrict__ dst, int n4) {
  int i = blockIdx.x * 256 + threadIdx.x;
  if (i >= n4) return;
  float4 v = ((const float4*)src)[i];
  ((ushort4*)dst)[i] = make_ushort4(f2bf(v.x), f2bf(v.y), f2bf(v.z), f2bf(v.w));
}

__global__ void fill_f32_k(float* __restrict__ p, float v, int n) {
  int i = blockIdx.x * 256 + threadIdx.x;
  if (i < n) p[i] = v;
}

// per-node attention scores: s_src[n,h] = <H[n,h,:], att_src[h,:]>, same for dst
__global__ void node_scores_k(const float* __restrict__ H,
                              const float* __restrict__ asrc, const float* __restrict__ adst,
                              float* __restrict__ ssrc, float* __restrict__ sdst) {
  int i = blockIdx.x * 256 + threadIdx.x;
  if (i >= N_NODES * 4) return;
  int n = i >> 2, h = i & 3;
  const float4* hp = (const float4*)(H + (size_t)n * 256 + h * 64);
  const float4* pa = (const float4*)(asrc + h * 64);
  const float4* pb = (const float4*)(adst + h * 64);
  float ss = 0.f, sd = 0.f;
#pragma unroll
  for (int j = 0; j < 16; ++j) {
    float4 x = hp[j], a = pa[j], b = pb[j];
    ss += x.x * a.x + x.y * a.y + x.z * a.z + x.w * a.w;
    sd += x.x * b.x + x.y * b.y + x.z * b.z + x.w * b.w;
  }
  ssrc[i] = ss; sdst[i] = sd;
}

__global__ void edge_max_k(const int* __restrict__ ei, const float* __restrict__ ssrc,
                           const float* __restrict__ sdst, float* __restrict__ nmax) {
  int i = blockIdx.x * 256 + threadIdx.x;
  if (i >= ETOT * 4) return;
  int e = i >> 2, h = i & 3;
  int s, d; edge_endpoints(ei, e, s, d);
  float lg = leaky(ssrc[s * 4 + h] + sdst[d * 4 + h]);
  atomicMaxF32(&nmax[d * 4 + h], lg);
}

__global__ void edge_expsum_k(const int* __restrict__ ei, const float* __restrict__ ssrc,
                              const float* __restrict__ sdst, const float* __restrict__ nmax,
                              float* __restrict__ nden) {
  int i = blockIdx.x * 256 + threadIdx.x;
  if (i >= ETOT * 4) return;
  int e = i >> 2, h = i & 3;
  int s, d; edge_endpoints(ei, e, s, d);
  float lg = leaky(ssrc[s * 4 + h] + sdst[d * 4 + h]);
  atomicAdd(&nden[d * 4 + h], __expf(lg - nmax[d * 4 + h]));
}

// one wave32 per edge (edges never straddle waves: 32 | wave base), 8 ch/lane.
// Lanes 0..3 compute the per-head alpha; every lane fetches its head's alpha
// via wave32 __shfl, avoiding 8x redundant scalar gathers per edge.
__global__ void edge_agg_k(const int* __restrict__ ei, const float* __restrict__ ssrc,
                           const float* __restrict__ sdst, const float* __restrict__ nmax,
                           const float* __restrict__ nden, const float* __restrict__ H,
                           float* __restrict__ OUT) {
  long long gid = (long long)blockIdx.x * 256 + threadIdx.x;
  if (gid >= (long long)ETOT * 32) return;
  int e = (int)(gid >> 5);
  int q = (int)(gid & 31);
  int s, d; edge_endpoints(ei, e, s, d);
  __builtin_prefetch(H + (size_t)s * 256 + (q << 3), 0, 0);  // global_prefetch_b8
  float alpha_l = 0.0f;
  if (q < 4) {
    float lg = leaky(ssrc[s * 4 + q] + sdst[d * 4 + q]);
    alpha_l = __expf(lg - nmax[d * 4 + q]) / nden[d * 4 + q];
  }
  int h = q >> 3;                       // head for this lane's 8 channels
  float alpha = __shfl(alpha_l, h, 32); // broadcast from lane h
  const float4* hp = (const float4*)(H + (size_t)s * 256 + (q << 3));
  float4 a0 = hp[0], a1 = hp[1];
  float* o = OUT + (size_t)d * 256 + (q << 3);
  atomicAdd(o + 0, a0.x * alpha);
  atomicAdd(o + 1, a0.y * alpha);
  atomicAdd(o + 2, a0.z * alpha);
  atomicAdd(o + 3, a0.w * alpha);
  atomicAdd(o + 4, a1.x * alpha);
  atomicAdd(o + 5, a1.y * alpha);
  atomicAdd(o + 6, a1.z * alpha);
  atomicAdd(o + 7, a1.w * alpha);
}

// relu(OUT + bias); to_bf16=1 -> emit bf16 for next GEMM; else keep f32 in place
__global__ void bias_act_k(float* __restrict__ OUT, const float* __restrict__ bias,
                           u16* __restrict__ Abf, int to_bf16) {
  int i = blockIdx.x * 256 + threadIdx.x;
  if (i >= N_NODES * 64) return;
  int n = i >> 6, c4 = (i & 63) << 2;
  float4 v = *(float4*)(OUT + (size_t)n * 256 + c4);
  const float4 b = *(const float4*)(bias + c4);
  v.x = fmaxf(v.x + b.x, 0.0f); v.y = fmaxf(v.y + b.y, 0.0f);
  v.z = fmaxf(v.z + b.z, 0.0f); v.w = fmaxf(v.w + b.w, 0.0f);
  if (to_bf16) {
    *(ushort4*)(Abf + (size_t)n * 256 + c4) =
        make_ushort4(f2bf(v.x), f2bf(v.y), f2bf(v.z), f2bf(v.w));
  } else {
    *(float4*)(OUT + (size_t)n * 256 + c4) = v;
  }
}

__global__ void pool_k(const float* __restrict__ P, const int* __restrict__ batch,
                       float* __restrict__ pooled) {
  int i = blockIdx.x * 256 + threadIdx.x;
  if (i >= N_NODES * 64) return;
  int n = i >> 6, c4 = (i & 63) << 2;
  int g = batch[n];
  float4 v = *(const float4*)(P + (size_t)n * 256 + c4);
  float* q = pooled + g * 256 + c4;
  atomicAdd(q + 0, v.x); atomicAdd(q + 1, v.y);
  atomicAdd(q + 2, v.z); atomicAdd(q + 3, v.w);
}

// one block per graph: fc1+bn+relu, fc2+bn+relu, out
__global__ __launch_bounds__(128) void mlp_head_k(
    const float* __restrict__ pooled,
    const float* fc1_w, const float* fc1_b,
    const float* g1, const float* be1, const float* m1, const float* v1,
    const float* fc2_w, const float* fc2_b,
    const float* g2, const float* be2, const float* m2, const float* v2,
    const float* ow, const float* ob, float* __restrict__ out) {
  __shared__ float sp[256];
  __shared__ float z1[128];
  __shared__ float z2[64];
  int g = blockIdx.x, t = threadIdx.x;
  sp[t] = pooled[g * 256 + t];
  sp[t + 128] = pooled[g * 256 + 128 + t];
  __syncthreads();
  float a = fc1_b[t];
  for (int k = 0; k < 256; ++k) a += sp[k] * fc1_w[k * 128 + t];
  a = (a - m1[t]) * rsqrtf(v1[t] + 1e-5f) * g1[t] + be1[t];
  z1[t] = fmaxf(a, 0.0f);
  __syncthreads();
  if (t < 64) {
    float b = fc2_b[t];
    for (int k = 0; k < 128; ++k) b += z1[k] * fc2_w[k * 64 + t];
    b = (b - m2[t]) * rsqrtf(v2[t] + 1e-5f) * g2[t] + be2[t];
    z2[t] = fmaxf(b, 0.0f);
  }
  __syncthreads();
  if (t == 0) {
    float o = ob[0];
    for (int k = 0; k < 64; ++k) o += z2[k] * ow[k];
    out[g] = o;
  }
}

// ---------------------------------------------------------------------------
extern "C" void kernel_launch(void* const* d_in, const int* in_sizes, int n_in,
                              void* d_out, int out_size, void* d_ws, size_t ws_size,
                              hipStream_t stream) {
  const float* x     = (const float*)d_in[0];
  const int*   ei    = (const int*)  d_in[1];
  const int*   batch = (const int*)  d_in[2];
  const float* W1    = (const float*)d_in[3];
  const float* as1   = (const float*)d_in[4];
  const float* ad1   = (const float*)d_in[5];
  const float* b1    = (const float*)d_in[6];
  const float* W2    = (const float*)d_in[7];
  const float* as2   = (const float*)d_in[8];
  const float* ad2   = (const float*)d_in[9];
  const float* b2    = (const float*)d_in[10];
  const float* fc1_w = (const float*)d_in[11];
  const float* fc1_b = (const float*)d_in[12];
  const float* bn1_g = (const float*)d_in[13];
  const float* bn1_b = (const float*)d_in[14];
  const float* bn1_m = (const float*)d_in[15];
  const float* bn1_v = (const float*)d_in[16];
  const float* fc2_w = (const float*)d_in[17];
  const float* fc2_b = (const float*)d_in[18];
  const float* bn2_g = (const float*)d_in[19];
  const float* bn2_b = (const float*)d_in[20];
  const float* bn2_m = (const float*)d_in[21];
  const float* bn2_v = (const float*)d_in[22];
  const float* out_w = (const float*)d_in[23];
  const float* out_b = (const float*)d_in[24];
  float* out = (float*)d_out;

  // workspace carve-up (all 16B-aligned offsets)
  char* p = (char*)d_ws;
  u16*   Abf    = (u16*)p;   p += (size_t)N_NODES * 256 * 2;   // 25.6 MB
  u16*   Wt     = (u16*)p;   p += (size_t)256 * 256 * 2;       // 131 KB (transposed weights)
  float* H      = (float*)p; p += (size_t)N_NODES * 256 * 4;   // 51.2 MB
  float* OUT    = (float*)p; p += (size_t)N_NODES * 256 * 4;   // 51.2 MB
  float* ssrc   = (float*)p; p += (size_t)N_NODES * 4 * 4;
  float* sdst   = (float*)p; p += (size_t)N_NODES * 4 * 4;
  float* nmax   = (float*)p; p += (size_t)N_NODES * 4 * 4;
  float* nden   = (float*)p; p += (size_t)N_NODES * 4 * 4;
  float* pooled = (float*)p; p += (size_t)GRAPHS * 256 * 4;

  const int B = 256;
  const int gN4   = (N_NODES * 4 + B - 1) / B;
  const int gN64  = (N_NODES * 64 + B - 1) / B;
  const int gE4   = (ETOT * 4 + B - 1) / B;
  const unsigned gE32 = (unsigned)(((long long)ETOT * 32 + B - 1) / B);

  // ===== Layer 1 =====
  cvt_bf16_k<<<(N_NODES * 128 / 4 + B - 1) / B, B, 0, stream>>>(x, Abf, N_NODES * 128 / 4);
  w_transpose_bf16_k<<<(128 * 256 + B - 1) / B, B, 0, stream>>>(W1, Wt, 128);
  gemm_bf16_wmma<<<N_NODES / 16, 128, 0, stream>>>(Abf, Wt, H, 128);
  node_scores_k<<<gN4, B, 0, stream>>>(H, as1, ad1, ssrc, sdst);
  fill_f32_k<<<(N_NODES * 256 + B - 1) / B, B, 0, stream>>>(OUT, 0.0f, N_NODES * 256);
  fill_f32_k<<<gN4, B, 0, stream>>>(nmax, -3.0e38f, N_NODES * 4);
  fill_f32_k<<<gN4, B, 0, stream>>>(nden, 0.0f, N_NODES * 4);
  edge_max_k<<<gE4, B, 0, stream>>>(ei, ssrc, sdst, nmax);
  edge_expsum_k<<<gE4, B, 0, stream>>>(ei, ssrc, sdst, nmax, nden);
  edge_agg_k<<<gE32, B, 0, stream>>>(ei, ssrc, sdst, nmax, nden, H, OUT);
  bias_act_k<<<gN64, B, 0, stream>>>(OUT, b1, Abf, 1);   // -> bf16 input of layer 2

  // ===== Layer 2 =====
  w_transpose_bf16_k<<<(256 * 256 + B - 1) / B, B, 0, stream>>>(W2, Wt, 256);
  gemm_bf16_wmma<<<N_NODES / 16, 128, 0, stream>>>(Abf, Wt, H, 256);
  node_scores_k<<<gN4, B, 0, stream>>>(H, as2, ad2, ssrc, sdst);
  fill_f32_k<<<(N_NODES * 256 + B - 1) / B, B, 0, stream>>>(OUT, 0.0f, N_NODES * 256);
  fill_f32_k<<<gN4, B, 0, stream>>>(nmax, -3.0e38f, N_NODES * 4);
  fill_f32_k<<<gN4, B, 0, stream>>>(nden, 0.0f, N_NODES * 4);
  edge_max_k<<<gE4, B, 0, stream>>>(ei, ssrc, sdst, nmax);
  edge_expsum_k<<<gE4, B, 0, stream>>>(ei, ssrc, sdst, nmax, nden);
  edge_agg_k<<<gE32, B, 0, stream>>>(ei, ssrc, sdst, nmax, nden, H, OUT);
  bias_act_k<<<gN64, B, 0, stream>>>(OUT, b2, Abf, 0);   // keep f32 in OUT

  // ===== Pool + MLP head =====
  fill_f32_k<<<(GRAPHS * 256 + B - 1) / B, B, 0, stream>>>(pooled, 0.0f, GRAPHS * 256);
  pool_k<<<gN64, B, 0, stream>>>(OUT, batch, pooled);
  mlp_head_k<<<GRAPHS, 128, 0, stream>>>(pooled, fc1_w, fc1_b,
                                         bn1_g, bn1_b, bn1_m, bn1_v,
                                         fc2_w, fc2_b, bn2_g, bn2_b, bn2_m, bn2_v,
                                         out_w, out_b, out);
}
